// GPT2LMHeadModel_5669356831060
// MI455X (gfx1250) — compile-verified, matrix operand order checked by reading, not run on hardware
//
#include <hip/hip_runtime.h>
#include <hip/hip_bf16.h>
#include <math.h>

// ---------------------------------------------------------------------------
// GPT-2 small forward pass for gfx1250 (MI455X), WMMA f16 (f32 accumulate).
// Round 3: transpose moved to the global-load side -> all LDS staging stores
// are packed ds_store_b64 (no more scalar b16 scatter).
// ---------------------------------------------------------------------------

#define BATCH 2
#define SEQ   1024
#define DMODEL 768
#define NHEAD 12
#define DHEAD 64
#define FFDIM 3072
#define NLAYER 6
#define VOCAB 50257
#define NTOK (BATCH * SEQ)   // 2048 rows in the activation matrices

typedef __attribute__((ext_vector_type(16))) _Float16 v16h;
typedef __attribute__((ext_vector_type(8)))  _Float16 v8h;
typedef __attribute__((ext_vector_type(4)))  _Float16 v4h;
typedef __attribute__((ext_vector_type(8)))  float    v8f;

static __device__ __forceinline__ v4h cvt4(const float4 f) {
    return (v4h){(_Float16)f.x, (_Float16)f.y, (_Float16)f.z, (_Float16)f.w};
}

// ============================ embedding ====================================
__global__ __launch_bounds__(256)
void embed_kernel(const int* __restrict__ ids, const float* __restrict__ wte,
                  const float* __restrict__ wpe, float* __restrict__ x)
{
    const int row = blockIdx.x;            // 0..NTOK-1
    const int s   = row & (SEQ - 1);       // position
    const int id  = ids[row];
    const float* wt = wte + (size_t)id * DMODEL;
    const float* wp = wpe + (size_t)s  * DMODEL;
    float* xr = x + (size_t)row * DMODEL;
    for (int i = threadIdx.x; i < DMODEL; i += 256)
        xr[i] = wt[i] + wp[i];
}

// ============================ layernorm ====================================
__global__ __launch_bounds__(256)
void ln_kernel(const float* __restrict__ x, const float* __restrict__ g,
               const float* __restrict__ b, float* __restrict__ out)
{
    __shared__ float red[256];
    const int row = blockIdx.x;
    const int tid = threadIdx.x;
    const float* xr = x + (size_t)row * DMODEL;

    float s = 0.f;
    for (int i = tid; i < DMODEL; i += 256) s += xr[i];
    red[tid] = s; __syncthreads();
    for (int off = 128; off > 0; off >>= 1) {
        if (tid < off) red[tid] += red[tid + off];
        __syncthreads();
    }
    const float mu = red[0] * (1.f / DMODEL);
    __syncthreads();

    float v = 0.f;
    for (int i = tid; i < DMODEL; i += 256) { float d = xr[i] - mu; v += d * d; }
    red[tid] = v; __syncthreads();
    for (int off = 128; off > 0; off >>= 1) {
        if (tid < off) red[tid] += red[tid + off];
        __syncthreads();
    }
    const float rstd = rsqrtf(red[0] * (1.f / DMODEL) + 1e-5f);

    float* orow = out + (size_t)row * DMODEL;
    for (int i = tid; i < DMODEL; i += 256)
        orow[i] = (xr[i] - mu) * rstd * g[i] + b[i];
}

// ============================ WMMA GEMM ====================================
// C[M,N] = epi( A[M,K] @ B + bias ).  A, B fp32 in global, converted to f16
// while staging to LDS.  BTRANS=false: B is W[K,N] row-major (loaded
// column-wise so LDS stores are packed).  BTRANS=true: B is Wt[N,K] row-major
// (logical B = Wt^T, used for x @ wte.T).
#define BM 128
#define BN 128
#define BK 32
#define LDSROW (BK + 8)    // 40 halves -> 80B row stride, keeps b128 alignment

enum { EPI_STORE = 0, EPI_GELU = 1, EPI_RES = 2 };

template<int EPI, bool BTRANS>
__global__ __launch_bounds__(256)
void gemm_kernel(const float* __restrict__ A, const float* __restrict__ Bw,
                 const float* __restrict__ bias, float* __restrict__ C,
                 const float* __restrict__ R, int M, int N, int K)
{
    __shared__ _Float16 As[BM * LDSROW];
    __shared__ _Float16 Bs[BN * LDSROW];

    const int tid  = threadIdx.x;
    const int lane = tid & 31;
    const int wave = tid >> 5;
    const int wm   = wave & 1;      // 0..1  : 64 rows each
    const int wn   = wave >> 1;     // 0..3  : 32 cols each
    const int hi   = lane >> 4;     // half-wave selector (K-halves)
    const int l15  = lane & 15;
    const int m0   = blockIdx.y * BM;
    const int n0   = blockIdx.x * BN;

    v8f acc[4][2];
#pragma unroll
    for (int i = 0; i < 4; ++i)
#pragma unroll
        for (int j = 0; j < 2; ++j)
            acc[i][j] = (v8f){0.f,0.f,0.f,0.f,0.f,0.f,0.f,0.f};

    float4 ra[4];           // register staging for the software pipeline
    float4 rb[4];           // BTRANS path (row-contiguous)
    float  rbc[4][4];       // non-trans path (column-wise gathers)

    auto loadA = [&](int k0) {
#pragma unroll
        for (int it = 0; it < 4; ++it) {
            const int idx = tid + it * 256;      // 1024 float4
            const int row = idx >> 3;            // 8 float4 per row
            const int c4  = (idx & 7) * 4;
            ra[it] = *(const float4*)(A + (size_t)(m0 + row) * K + k0 + c4);
        }
    };
    auto loadB = [&](int k0) {
        if (!BTRANS) {
            // column-wise: 4 consecutive K of one column per group
#pragma unroll
            for (int g = 0; g < 4; ++g) {
                const int idx = tid + g * 256;   // 1024 groups
                const int n   = idx & 127;
                const int kq  = (idx >> 7) * 4;  // 0,4,..,28
#pragma unroll
                for (int j = 0; j < 4; ++j)
                    rbc[g][j] = Bw[(size_t)(k0 + kq + j) * N + n0 + n];
            }
        } else {
#pragma unroll
            for (int it = 0; it < 4; ++it) {
                const int idx = tid + it * 256;
                const int nr  = idx >> 3;        // 8 float4 per n-row
                const int c4  = (idx & 7) * 4;
                rb[it] = make_float4(0.f, 0.f, 0.f, 0.f);
                if (n0 + nr < N)
                    rb[it] = *(const float4*)(Bw + (size_t)(n0 + nr) * K + k0 + c4);
            }
        }
    };
    auto stageAB = [&]() {
#pragma unroll
        for (int it = 0; it < 4; ++it) {         // A: packed b64 stores
            const int idx = tid + it * 256;
            const int row = idx >> 3;
            const int c4  = (idx & 7) * 4;
            *(v4h*)&As[row * LDSROW + c4] = cvt4(ra[it]);
        }
        if (!BTRANS) {                           // B: packed b64 stores too
#pragma unroll
            for (int g = 0; g < 4; ++g) {
                const int idx = tid + g * 256;
                const int n   = idx & 127;
                const int kq  = (idx >> 7) * 4;
                *(v4h*)&Bs[n * LDSROW + kq] =
                    cvt4(make_float4(rbc[g][0], rbc[g][1], rbc[g][2], rbc[g][3]));
            }
        } else {                                 // Bt rows are contiguous: b64
#pragma unroll
            for (int it = 0; it < 4; ++it) {
                const int idx = tid + it * 256;
                const int nr  = idx >> 3;
                const int c4  = (idx & 7) * 4;
                *(v4h*)&Bs[nr * LDSROW + c4] = cvt4(rb[it]);
            }
        }
    };

    loadA(0); loadB(0);
    for (int k0 = 0; k0 < K; k0 += BK) {
        __syncthreads();                 // previous tile fully consumed
        stageAB();
        __syncthreads();                 // tile visible to all waves
        if (k0 + BK < K) { loadA(k0 + BK); loadB(k0 + BK); }  // overlap w/ WMMA

        v16h afrag[4];
#pragma unroll
        for (int ms = 0; ms < 4; ++ms) {
            const _Float16* p = &As[(wm * 64 + ms * 16 + l15) * LDSROW + hi * 8];
            const v8h lo = *(const v8h*)p;
            const v8h hb = *(const v8h*)(p + 16);
#pragma unroll
            for (int i = 0; i < 8; ++i) { afrag[ms][i] = lo[i]; afrag[ms][i + 8] = hb[i]; }
        }
#pragma unroll
        for (int ns = 0; ns < 2; ++ns) {
            const _Float16* p = &Bs[(wn * 32 + ns * 16 + l15) * LDSROW + hi * 8];
            const v8h lo = *(const v8h*)p;
            const v8h hb = *(const v8h*)(p + 16);
            v16h bfrag;
#pragma unroll
            for (int i = 0; i < 8; ++i) { bfrag[i] = lo[i]; bfrag[i + 8] = hb[i]; }
#pragma unroll
            for (int ms = 0; ms < 4; ++ms)
                acc[ms][ns] = __builtin_amdgcn_wmma_f32_16x16x32_f16(
                    false, afrag[ms], false, bfrag, (short)0, acc[ms][ns], false, false);
        }
    }

    // ---- epilogue ----
#pragma unroll
    for (int ms = 0; ms < 4; ++ms)
#pragma unroll
        for (int ns = 0; ns < 2; ++ns) {
            const int col = n0 + wn * 32 + ns * 16 + l15;
            if (col >= N) continue;
            const float bv = bias ? bias[col] : 0.f;
#pragma unroll
            for (int r = 0; r < 8; ++r) {
                const int row = m0 + wm * 64 + ms * 16 + r + hi * 8;
                float val = acc[ms][ns][r] + bv;
                const size_t off = (size_t)row * N + col;
                if constexpr (EPI == EPI_GELU)
                    val = 0.5f * val * (1.f + erff(val * 0.70710678118654752f));
                if constexpr (EPI == EPI_RES)
                    C[off] = R[off] + val;
                else
                    C[off] = val;
            }
        }
}

// ========================= flash-style attention ===========================
// grid (S/64, H, B); 128 threads = 4 waves, wave w owns query rows w*16..+15.
__global__ __launch_bounds__(128)
void attn_kernel(const float* __restrict__ q, const float* __restrict__ k,
                 const float* __restrict__ v, float* __restrict__ o)
{
    __shared__ _Float16 Qs[64 * 72];        // [qrow][dh], stride 72
    __shared__ _Float16 Ks[32 * 72];        // [key ][dh], stride 72
    __shared__ _Float16 Vt[64 * 40];        // [dh  ][key], stride 40
    __shared__ _Float16 Ps[4 * 16 * 40];    // per-wave P staging [row][key]

    const int qt = blockIdx.x, hh = blockIdx.y, bb = blockIdx.z;
    const int tid = threadIdx.x, lane = tid & 31, wave = tid >> 5;
    const int hi = lane >> 4, l15 = lane & 15;
    const size_t rowBase = (size_t)bb * SEQ + qt * 64;   // into [B*S, D]
    const int colBase = hh * DHEAD;

    // ---- load Q tile (scaled by 1/sqrt(DH)), packed f16 stores ----
#pragma unroll
    for (int it = 0; it < 8; ++it) {
        const int idx = tid + it * 128;   // 1024 float4
        const int row = idx >> 4;         // 16 float4 per 64-wide row
        const int c4  = (idx & 15) * 4;
        float4 f = *(const float4*)(q + (rowBase + row) * DMODEL + colBase + c4);
        f.x *= 0.125f; f.y *= 0.125f; f.z *= 0.125f; f.w *= 0.125f;
        *(v4h*)&Qs[row * 72 + c4] = cvt4(f);
    }
    __syncthreads();

    // ---- preload Q A-fragments (K-dim = 64 -> 2 fragments) ----
    v16h aq[2];
#pragma unroll
    for (int ks = 0; ks < 2; ++ks) {
        const _Float16* p = &Qs[(wave * 16 + l15) * 72 + ks * 32 + hi * 8];
        const v8h lo = *(const v8h*)p;
        const v8h hb = *(const v8h*)(p + 16);
#pragma unroll
        for (int i = 0; i < 8; ++i) { aq[ks][i] = lo[i]; aq[ks][i + 8] = hb[i]; }
    }

    float m_i[8], l_i[8];
    v8f ao[4];
#pragma unroll
    for (int r = 0; r < 8; ++r) { m_i[r] = -1e30f; l_i[r] = 0.f; }
#pragma unroll
    for (int d = 0; d < 4; ++d) ao[d] = (v8f){0.f,0.f,0.f,0.f,0.f,0.f,0.f,0.f};

    const int ktmax = (qt * 64 + 63) >> 5;     // inclusive; block-uniform
    for (int kt = 0; kt <= ktmax; ++kt) {
        __syncthreads();
        const size_t keyBase = (size_t)(bb * SEQ + kt * 32);
        // ---- stage K row-major (packed b128 load -> packed b64 store) ----
#pragma unroll
        for (int it = 0; it < 4; ++it) {
            const int idx = tid + it * 128;   // 512 float4
            const int row = idx >> 4;         // 0..31 keys
            const int c4  = (idx & 15) * 4;
            const float4 fk = *(const float4*)(k + (keyBase + row) * DMODEL + colBase + c4);
            *(v4h*)&Ks[row * 72 + c4] = cvt4(fk);
        }
        // ---- stage V transposed: column-wise gathers -> packed b64 store ----
#pragma unroll
        for (int it = 0; it < 4; ++it) {
            const int idx = tid + it * 128;   // 512 groups of 4 keys
            const int dh  = idx & 63;
            const int kq  = (idx >> 6) * 4;   // 0,4,..,28
            float t[4];
#pragma unroll
            for (int j = 0; j < 4; ++j)
                t[j] = v[(keyBase + kq + j) * DMODEL + colBase + dh];
            *(v4h*)&Vt[dh * 40 + kq] = cvt4(make_float4(t[0], t[1], t[2], t[3]));
        }
        __syncthreads();

        // ---- scores = Q x K^T : 16x32 per wave ----
        v8f sc[2];
#pragma unroll
        for (int ns = 0; ns < 2; ++ns) {
            sc[ns] = (v8f){0.f,0.f,0.f,0.f,0.f,0.f,0.f,0.f};
#pragma unroll
            for (int ks = 0; ks < 2; ++ks) {
                const _Float16* p = &Ks[(ns * 16 + l15) * 72 + ks * 32 + hi * 8];
                const v8h lo = *(const v8h*)p;
                const v8h hb = *(const v8h*)(p + 16);
                v16h bfrag;
#pragma unroll
                for (int i = 0; i < 8; ++i) { bfrag[i] = lo[i]; bfrag[i + 8] = hb[i]; }
                sc[ns] = __builtin_amdgcn_wmma_f32_16x16x32_f16(
                    false, aq[ks], false, bfrag, (short)0, sc[ns], false, false);
            }
        }

        // ---- causal mask + online softmax ----
        float alpha[8];
#pragma unroll
        for (int r = 0; r < 8; ++r) {
            const int qrow = qt * 64 + wave * 16 + r + hi * 8;
            float mx = m_i[r];
#pragma unroll
            for (int ns = 0; ns < 2; ++ns) {
                const int key = kt * 32 + ns * 16 + l15;
                float s = (key <= qrow) ? sc[ns][r] : -1e30f;
                sc[ns][r] = s;
                mx = fmaxf(mx, s);
            }
#pragma unroll
            for (int msk = 1; msk < 16; msk <<= 1)
                mx = fmaxf(mx, __shfl_xor(mx, msk, 32));
            alpha[r] = __expf(m_i[r] - mx);
            float rs = 0.f;
#pragma unroll
            for (int ns = 0; ns < 2; ++ns) {
                const float p = __expf(sc[ns][r] - mx);
                sc[ns][r] = p; rs += p;
            }
#pragma unroll
            for (int msk = 1; msk < 16; msk <<= 1)
                rs += __shfl_xor(rs, msk, 32);
            l_i[r] = l_i[r] * alpha[r] + rs;
            m_i[r] = mx;
        }
#pragma unroll
        for (int d = 0; d < 4; ++d)
#pragma unroll
            for (int r = 0; r < 8; ++r) ao[d][r] *= alpha[r];

        // ---- stage P (C-layout -> A-layout) through LDS ----
        _Float16* pw = &Ps[wave * 16 * 40];
#pragma unroll
        for (int ns = 0; ns < 2; ++ns)
#pragma unroll
            for (int r = 0; r < 8; ++r)
                pw[(r + hi * 8) * 40 + ns * 16 + l15] = (_Float16)sc[ns][r];
        __syncthreads();

        v16h pf;
        {
            const _Float16* p = &Ps[wave * 16 * 40 + l15 * 40 + hi * 8];
            const v8h lo = *(const v8h*)p;
            const v8h hb = *(const v8h*)(p + 16);
#pragma unroll
            for (int i = 0; i < 8; ++i) { pf[i] = lo[i]; pf[i + 8] = hb[i]; }
        }
        // ---- O += P x V ----
#pragma unroll
        for (int d = 0; d < 4; ++d) {
            const _Float16* p = &Vt[(d * 16 + l15) * 40 + hi * 8];
            const v8h lo = *(const v8h*)p;
            const v8h hb = *(const v8h*)(p + 16);
            v16h bfrag;
#pragma unroll
            for (int i = 0; i < 8; ++i) { bfrag[i] = lo[i]; bfrag[i + 8] = hb[i]; }
            ao[d] = __builtin_amdgcn_wmma_f32_16x16x32_f16(
                false, pf, false, bfrag, (short)0, ao[d], false, false);
        }
    }

    // ---- write O (normalized) ----
#pragma unroll
    for (int r = 0; r < 8; ++r) {
        const float inv = 1.f / l_i[r];
        const size_t row = rowBase + wave * 16 + r + hi * 8;
#pragma unroll
        for (int d = 0; d < 4; ++d) {
            const int col = colBase + d * 16 + l15;
            o[row * DMODEL + col] = ao[d][r] * inv;
        }
    }
}

// ============================ launcher =====================================
extern "C" void kernel_launch(void* const* d_in, const int* in_sizes, int n_in,
                              void* d_out, int out_size, void* d_ws, size_t ws_size,
                              hipStream_t stream)
{
    const int*   ids   = (const int*)  d_in[0];
    const float* wte   = (const float*)d_in[1];
    const float* wpe   = (const float*)d_in[2];
    const float* Wq    = (const float*)d_in[3];
    const float* bq    = (const float*)d_in[4];
    const float* Wk    = (const float*)d_in[5];
    const float* bk    = (const float*)d_in[6];
    const float* Wv    = (const float*)d_in[7];
    const float* bv    = (const float*)d_in[8];
    const float* Wo    = (const float*)d_in[9];
    const float* bo    = (const float*)d_in[10];
    const float* Wfc   = (const float*)d_in[11];
    const float* bfc   = (const float*)d_in[12];
    const float* Wproj = (const float*)d_in[13];
    const float* bproj = (const float*)d_in[14];
    const float* ln1g  = (const float*)d_in[15];
    const float* ln1b  = (const float*)d_in[16];
    const float* lnfg  = (const float*)d_in[17];
    const float* lnfb  = (const float*)d_in[18];
    float* logits = (float*)d_out;

    // workspace carve-up (all fp32): ~63 MB
    const size_t NM = (size_t)NTOK * DMODEL;
    float* x  = (float*)d_ws;
    float* h  = x  + NM;
    float* qb = h  + NM;
    float* kb = qb + NM;
    float* vb = kb + NM;
    float* ob = vb + NM;
    float* ff = ob + NM;

    embed_kernel<<<NTOK, 256, 0, stream>>>(ids, wte, wpe, x);

    const dim3 gDD(DMODEL / BN, NTOK / BM);     // 6 x 16
    const dim3 gDF(FFDIM  / BN, NTOK / BM);     // 24 x 16
    const dim3 gAttn(SEQ / 64, NHEAD, BATCH);   // 16 x 12 x 2

    for (int l = 0; l < NLAYER; ++l) {
        ln_kernel<<<NTOK, 256, 0, stream>>>(x, ln1g + l * DMODEL, ln1b + l * DMODEL, h);

        gemm_kernel<EPI_STORE, false><<<gDD, 256, 0, stream>>>(
            h, Wq + (size_t)l * DMODEL * DMODEL, bq + l * DMODEL, qb, nullptr,
            NTOK, DMODEL, DMODEL);
        gemm_kernel<EPI_STORE, false><<<gDD, 256, 0, stream>>>(
            h, Wk + (size_t)l * DMODEL * DMODEL, bk + l * DMODEL, kb, nullptr,
            NTOK, DMODEL, DMODEL);
        gemm_kernel<EPI_STORE, false><<<gDD, 256, 0, stream>>>(
            h, Wv + (size_t)l * DMODEL * DMODEL, bv + l * DMODEL, vb, nullptr,
            NTOK, DMODEL, DMODEL);

        attn_kernel<<<gAttn, 128, 0, stream>>>(qb, kb, vb, ob);

        gemm_kernel<EPI_RES, false><<<gDD, 256, 0, stream>>>(
            ob, Wo + (size_t)l * DMODEL * DMODEL, bo + l * DMODEL, x, x,
            NTOK, DMODEL, DMODEL);

        // NOTE: reference applies MLP to the residual directly (no ln_2)
        gemm_kernel<EPI_GELU, false><<<gDF, 256, 0, stream>>>(
            x, Wfc + (size_t)l * DMODEL * FFDIM, bfc + l * FFDIM, ff, nullptr,
            NTOK, FFDIM, DMODEL);
        gemm_kernel<EPI_RES, false><<<gDD, 256, 0, stream>>>(
            ff, Wproj + (size_t)l * FFDIM * DMODEL, bproj + l * DMODEL, x, x,
            NTOK, DMODEL, FFDIM);
    }

    ln_kernel<<<NTOK, 256, 0, stream>>>(x, lnfg, lnfb, h);

    const dim3 gLM((VOCAB + BN - 1) / BN, NTOK / BM);   // 393 x 16
    gemm_kernel<EPI_STORE, true><<<gLM, 256, 0, stream>>>(
        h, wte, nullptr, logits, nullptr, NTOK, VOCAB, DMODEL);
}
